// MultiHeadAttention_25254407701065
// MI455X (gfx1250) — matrix operand order repaired and checked by reading, not compiled, not run on hardware
//
#include <hip/hip_runtime.h>

// ---------------------------------------------------------------------------
// MultiHeadAttention forward, MI455X (gfx1250, wave32, WMMA bf16).
// B=4, T=2048, E=1024, H=16, D=64.
// All matmuls via v_wmma_f32_16x16x32_bf16 (bf16 in, fp32 accumulate).
// ---------------------------------------------------------------------------

constexpr int B = 4;
constexpr int T = 2048;
constexpr int E = 1024;
constexpr int H = 16;
constexpr int D = 64;

typedef __attribute__((ext_vector_type(16))) __bf16          v16bf;
typedef __attribute__((ext_vector_type(8)))  float           v8f;
typedef __attribute__((ext_vector_type(16))) unsigned short  u16x16;
typedef __attribute__((ext_vector_type(8)))  unsigned short  u16x8;

// float -> bf16 (round to nearest even)
__device__ __forceinline__ unsigned short f2bf(float f) {
    union { float f; unsigned int u; } cv; cv.f = f;
    unsigned int u = cv.u;
    unsigned int r = u + 0x7fffu + ((u >> 16) & 1u);
    return (unsigned short)(r >> 16);
}

// A-operand loader: 16x32 bf16 tile, this lane = row (lane&15), hh = lane>>4.
// Element i: k = hh*16 + i.
__device__ __forceinline__ v16bf loadA(const unsigned short* p, int hh) {
    u16x8 lo = *reinterpret_cast<const u16x8*>(p + hh * 16);
    u16x8 hi = *reinterpret_cast<const u16x8*>(p + hh * 16 + 8);
    u16x16 v = __builtin_shufflevector(lo, hi, 0,1,2,3,4,5,6,7,8,9,10,11,12,13,14,15);
    return __builtin_bit_cast(v16bf, v);
}

// B-operand loader: 32x16 bf16 tile, this lane = col (lane&15), hh = lane>>4.
// Element i: k = hh*16 + i -> caller passes p = col base + hh*16 (32B aligned).
__device__ __forceinline__ v16bf loadB(const unsigned short* p) {
    u16x16 v = *reinterpret_cast<const u16x16*>(p);
    return __builtin_bit_cast(v16bf, v);
}

__device__ __forceinline__ v8f wmma_bf16(v16bf a, v16bf b, v8f c) {
    return __builtin_amdgcn_wmma_f32_16x16x32_bf16(false, a, false, b,
                                                   (short)0, c, false, false);
}

// ---------------------------------------------------------------------------
// Elementwise fp32 -> bf16 convert
// ---------------------------------------------------------------------------
__global__ void k_convert(const float* __restrict__ src,
                          unsigned short* __restrict__ dst, int n) {
    int i = blockIdx.x * 256 + threadIdx.x;
    if (i < n) dst[i] = f2bf(src[i]);
}

// W[H,E,D] fp32 -> Wt[H,D,E] bf16 (so GEMM B-loads are contiguous along E)
__global__ void k_transpose_w(const float* __restrict__ W,
                              unsigned short* __restrict__ Wt) {
    int idx = blockIdx.x * 256 + threadIdx.x;      // < H*E*D
    int h = idx / (E * D);
    int rem = idx - h * (E * D);
    int e = rem / D;
    int d = rem - e * D;
    Wt[((size_t)h * D + d) * E + e] = f2bf(W[idx]);
}

// ---------------------------------------------------------------------------
// Fused QKV projection: per wave, one 16(t) x 32(d) tile for each of Q, K, V
// of one (b,h): 6 accumulators, one shared A-load feeds 6 WMMAs per K-step.
// Q,K stored [B,H,T,D] bf16; V stored transposed [B,H,D,T] bf16.
// ---------------------------------------------------------------------------
__global__ void k_qkv(const unsigned short* __restrict__ xbf,
                      const unsigned short* __restrict__ wqt,
                      const unsigned short* __restrict__ wkt,
                      const unsigned short* __restrict__ wvt,
                      unsigned short* __restrict__ Qbf,
                      unsigned short* __restrict__ Kbf,
                      unsigned short* __restrict__ Vtb) {
    int lane = threadIdx.x & 31;
    int w = blockIdx.x * 8 + (threadIdx.x >> 5);
    int dt = w & 1;               // d half (2 x 32 cols)
    int tt = (w >> 1) & 127;      // t tile (T/16 = 128)
    int bh = w >> 8;              // b*H + h
    int h  = bh & 15;
    int mn = lane & 15;           // A-row / B-col index
    int hh = lane >> 4;
    int b  = bh >> 4;
    int d0 = dt * 32;

    const unsigned short* arow = xbf + ((size_t)(b * T + tt * 16 + mn)) * E;
    size_t wrow = ((size_t)(h * D + d0 + mn)) * E + hh * 16;
    const unsigned short* bq = wqt + wrow;   // col tile d0..d0+15; +16*E for next
    const unsigned short* bk = wkt + wrow;
    const unsigned short* bv = wvt + wrow;
    const size_t colstep = (size_t)16 * E;

    v8f cq0 = {}, cq1 = {}, ck0 = {}, ck1 = {}, cv0 = {}, cv1 = {};
    for (int e0 = 0; e0 < E; e0 += 32) {
        v16bf a = loadA(arow + e0, hh);
        cq0 = wmma_bf16(a, loadB(bq + e0),           cq0);
        cq1 = wmma_bf16(a, loadB(bq + colstep + e0), cq1);
        ck0 = wmma_bf16(a, loadB(bk + e0),           ck0);
        ck1 = wmma_bf16(a, loadB(bk + colstep + e0), ck1);
        cv0 = wmma_bf16(a, loadB(bv + e0),           cv0);
        cv1 = wmma_bf16(a, loadB(bv + colstep + e0), cv1);
    }

    for (int r = 0; r < 8; ++r) {
        int row = tt * 16 + r + 8 * hh;   // local t index
        size_t qk = ((size_t)bh * T + row) * D + d0 + mn;
        Qbf[qk]      = f2bf(cq0[r]);
        Qbf[qk + 16] = f2bf(cq1[r]);
        Kbf[qk]      = f2bf(ck0[r]);
        Kbf[qk + 16] = f2bf(ck1[r]);
        Vtb[((size_t)bh * D + d0 + mn) * T + row]        = f2bf(cv0[r]);
        Vtb[((size_t)bh * D + d0 + 16 + mn) * T + row]   = f2bf(cv1[r]);
    }
}

// ---------------------------------------------------------------------------
// Flash attention: per wave, one 16-query tile of one (b,h). Key chunks of 32
// (two 16-key score tiles), causal + padding mask, online softmax (fp32),
// P transposed through per-wave LDS slab into WMMA A-layout, then P*V.
// Q A-operands are loop-invariant and held in registers across all chunks.
// Output written bf16 to O[B*T, E] with E = h*64 + d ordering.
// ---------------------------------------------------------------------------
__global__ void k_attn(const unsigned short* __restrict__ Qbf,
                       const unsigned short* __restrict__ Kbf,
                       const unsigned short* __restrict__ Vtb,
                       const float* __restrict__ am,
                       unsigned short* __restrict__ Obf) {
    __shared__ __align__(64) unsigned short pbuf[8 * 16 * 32];  // 8 waves * 1KB
    int lane = threadIdx.x & 31;
    int wl = threadIdx.x >> 5;
    int w = blockIdx.x * 8 + wl;
    int qt = w & 127;             // query tile (T/16 = 128)
    int bh = w >> 7;
    int b  = bh >> 4;
    int q0 = qt * 16;
    int mn = lane & 15;
    int hh = lane >> 4;
    unsigned short* pb = pbuf + wl * 512;

    const unsigned short* qrow = Qbf + ((size_t)bh * T + q0 + mn) * D;
    const size_t kbase = (size_t)bh * T * D;
    const size_t vbase = (size_t)bh * D * T;
    const float NEG = -3.0e38f;

    // Q tile A-operands: invariant across key chunks, keep in registers
    v16bf qa0 = loadA(qrow, hh);
    v16bf qa1 = loadA(qrow + 32, hh);

    v8f acc[4] = {};
    float m_i[8], l_i[8];
    for (int r = 0; r < 8; ++r) { m_i[r] = -1.0e30f; l_i[r] = 0.0f; }

    int send = q0 + 16;
    for (int ks = 0; ks < send; ks += 32) {
        const unsigned short* k0p = Kbf + kbase + (size_t)(ks + mn) * D + hh * 16;
        const unsigned short* k1p = Kbf + kbase + (size_t)(ks + 16 + mn) * D + hh * 16;
        v8f sc0 = {}, sc1 = {};
        sc0 = wmma_bf16(qa0, loadB(k0p),      sc0);
        sc1 = wmma_bf16(qa0, loadB(k1p),      sc1);
        sc0 = wmma_bf16(qa1, loadB(k0p + 32), sc0);
        sc1 = wmma_bf16(qa1, loadB(k1p + 32), sc1);

        int key0 = ks + mn, key1 = ks + 16 + mn;
        float pad0 = (1.0f - am[(size_t)b * T + key0]) * -3.4028235e38f;
        float pad1 = (1.0f - am[(size_t)b * T + key1]) * -3.4028235e38f;

        for (int r = 0; r < 8; ++r) {
            int q = q0 + r + 8 * hh;
            float v0 = (key0 <= q) ? fmaxf(sc0[r] * 0.125f + pad0, NEG) : NEG;
            float v1 = (key1 <= q) ? fmaxf(sc1[r] * 0.125f + pad1, NEG) : NEG;
            // row max across the 16 lanes holding this row
            float mx = fmaxf(v0, v1);
            mx = fmaxf(mx, __shfl_xor(mx, 1));
            mx = fmaxf(mx, __shfl_xor(mx, 2));
            mx = fmaxf(mx, __shfl_xor(mx, 4));
            mx = fmaxf(mx, __shfl_xor(mx, 8));
            float nm = fmaxf(m_i[r], mx);
            float scale = __expf(m_i[r] - nm);
            m_i[r] = nm;
            float p0 = __expf(v0 - nm);
            float p1 = __expf(v1 - nm);
            float rs = p0 + p1;
            rs += __shfl_xor(rs, 1);
            rs += __shfl_xor(rs, 2);
            rs += __shfl_xor(rs, 4);
            rs += __shfl_xor(rs, 8);
            l_i[r] = l_i[r] * scale + rs;
            acc[0][r] *= scale; acc[1][r] *= scale;
            acc[2][r] *= scale; acc[3][r] *= scale;
            // store P (C/D layout -> row-major LDS) for A-layout reload
            pb[(r + 8 * hh) * 32 + mn]      = f2bf(p0);
            pb[(r + 8 * hh) * 32 + 16 + mn] = f2bf(p1);
        }
        asm volatile("s_wait_dscnt 0" ::: "memory");
        v16bf pa = loadA(pb + mn * 32, hh);
        for (int t = 0; t < 4; ++t) {
            const unsigned short* vcol =
                Vtb + vbase + (size_t)(t * 16 + mn) * T + ks + hh * 16;
            acc[t] = wmma_bf16(pa, loadB(vcol), acc[t]);
        }
        asm volatile("" ::: "memory");
    }

    for (int t = 0; t < 4; ++t)
        for (int r = 0; r < 8; ++r) {
            float o = acc[t][r] / l_i[r];
            Obf[((size_t)b * T + q0 + r + 8 * hh) * E + (bh & 15) * D + t * 16 + mn]
                = f2bf(o);
        }
}

// ---------------------------------------------------------------------------
// Output projection: out[B*T,E] = O[B*T,E] @ Wo^T + bo (fp32 out).
// Per wave: 16 rows x 64 cols (4 accumulators); one A-load feeds 4 WMMAs.
// B-operand = Wo[eo][e] row-major -> contiguous along k=e.
// ---------------------------------------------------------------------------
__global__ void k_oproj(const unsigned short* __restrict__ Obf,
                        const unsigned short* __restrict__ Wob,
                        const float* __restrict__ bo,
                        float* __restrict__ out) {
    int lane = threadIdx.x & 31;
    int w = blockIdx.x * 8 + (threadIdx.x >> 5);
    int nt = w & 15;              // E/64 = 16 output-col groups
    int mt = w >> 4;              // B*T/16 = 512 row tiles
    int mn = lane & 15;
    int hh = lane >> 4;

    const unsigned short* arow = Obf + ((size_t)(mt * 16 + mn)) * E;
    const unsigned short* brow = Wob + ((size_t)(nt * 64 + mn)) * E + hh * 16;
    const size_t colstep = (size_t)16 * E;

    v8f c[4] = {};
    for (int e0 = 0; e0 < E; e0 += 32) {
        v16bf a = loadA(arow + e0, hh);
        c[0] = wmma_bf16(a, loadB(brow + e0),               c[0]);
        c[1] = wmma_bf16(a, loadB(brow + colstep + e0),     c[1]);
        c[2] = wmma_bf16(a, loadB(brow + 2 * colstep + e0), c[2]);
        c[3] = wmma_bf16(a, loadB(brow + 3 * colstep + e0), c[3]);
    }

    for (int j = 0; j < 4; ++j) {
        float bias = bo[nt * 64 + j * 16 + mn];
        for (int r = 0; r < 8; ++r)
            out[((size_t)(mt * 16 + r + 8 * hh)) * E + nt * 64 + j * 16 + mn]
                = c[j][r] + bias;
    }
}

// ---------------------------------------------------------------------------
// Host launcher
// ---------------------------------------------------------------------------
extern "C" void kernel_launch(void* const* d_in, const int* in_sizes, int n_in,
                              void* d_out, int out_size, void* d_ws, size_t ws_size,
                              hipStream_t stream) {
    const float* x  = (const float*)d_in[0];
    const float* am = (const float*)d_in[1];
    const float* Wq = (const float*)d_in[2];
    const float* Wk = (const float*)d_in[3];
    const float* Wv = (const float*)d_in[4];
    const float* Wo = (const float*)d_in[5];
    const float* bo = (const float*)d_in[6];

    unsigned short* ws = (unsigned short*)d_ws;
    size_t o = 0;
    unsigned short* xbf = ws + o; o += (size_t)B * T * E;      // 8M elems
    unsigned short* Qbf = ws + o; o += (size_t)B * H * T * D;  // 8M
    unsigned short* Kbf = ws + o; o += (size_t)B * H * T * D;  // 8M
    unsigned short* Vtb = ws + o; o += (size_t)B * H * T * D;  // 8M
    unsigned short* Obf = ws + o; o += (size_t)B * T * E;      // 8M
    unsigned short* Wqt = ws + o; o += (size_t)H * E * D;      // 1M
    unsigned short* Wkt = ws + o; o += (size_t)H * E * D;      // 1M
    unsigned short* Wvt = ws + o; o += (size_t)H * E * D;      // 1M
    unsigned short* Wob = ws + o; o += (size_t)E * E;          // 1M
    // total: ~88 MB of workspace

    k_convert<<<(B * T * E) / 256, 256, 0, stream>>>(x, xbf, B * T * E);
    k_transpose_w<<<(H * E * D) / 256, 256, 0, stream>>>(Wq, Wqt);
    k_transpose_w<<<(H * E * D) / 256, 256, 0, stream>>>(Wk, Wkt);
    k_transpose_w<<<(H * E * D) / 256, 256, 0, stream>>>(Wv, Wvt);
    k_convert<<<(E * E) / 256, 256, 0, stream>>>(Wo, Wob, E * E);

    // B*H * (T/16) * (D/32) = 16384 waves -> 2048 blocks of 8 waves
    k_qkv<<<2048, 256, 0, stream>>>(xbf, Wqt, Wkt, Wvt, Qbf, Kbf, Vtb);
    // B*H * (T/16) = 8192 waves -> 1024 blocks
    k_attn<<<1024, 256, 0, stream>>>(Qbf, Kbf, Vtb, am, Obf);
    // (B*T/16) * (E/64) = 8192 waves -> 1024 blocks
    k_oproj<<<1024, 256, 0, stream>>>(Obf, Wob, bo, (float*)d_out);
}
